// RoPESelfAttention_31688268710537
// MI455X (gfx1250) — compile-verified, matrix operand order checked by reading, not run on hardware
//
#include <hip/hip_runtime.h>

#define BB 8
#define NN 1024
#define DD 768
#define HH 12
#define HD 64
#define TD 2304

typedef __attribute__((ext_vector_type(16))) __bf16 v16bf;
typedef __attribute__((ext_vector_type(8)))  __bf16 v8bf;
typedef __attribute__((ext_vector_type(8)))  float  v8f;

union FragU { v16bf v; v8bf h[2]; unsigned short us[16]; };

__device__ __forceinline__ unsigned short f32_to_bf16(float f) {
  __bf16 h = (__bf16)f;                       // native cvt, RNE
  return __builtin_bit_cast(unsigned short, h);
}
__device__ __forceinline__ float bf16_to_f32(unsigned short s) {
  return __uint_as_float(((unsigned int)s) << 16);
}

__device__ __forceinline__ v8f wmma_bf16(v16bf a, v16bf b, v8f c) {
  return __builtin_amdgcn_wmma_f32_16x16x32_bf16(false, a, false, b, (short)0, c,
                                                 false, false);
}

__device__ __forceinline__ v16bf frag_from_bf16(const unsigned short* p0,
                                                const unsigned short* p1) {
  FragU f;
  f.h[0] = *(const v8bf*)p0;
  f.h[1] = *(const v8bf*)p1;
  return f.v;
}

// ---------------------------------------------------------------------------
// Kernel 0: bulk f32 -> bf16 conversion (one pass, packed cvt)
// ---------------------------------------------------------------------------
__global__ void cvt_bf16_kernel(const float* __restrict__ src,
                                unsigned short* __restrict__ dst, int n4) {
  int i = blockIdx.x * blockDim.x + threadIdx.x;
  if (i < n4) {
    float4 v = ((const float4*)src)[i];
    ushort4 o;
    o.x = f32_to_bf16(v.x);
    o.y = f32_to_bf16(v.y);
    o.z = f32_to_bf16(v.z);
    o.w = f32_to_bf16(v.w);
    ((ushort4*)dst)[i] = o;
  }
}

// ---------------------------------------------------------------------------
// Kernel 1: qkv = x @ qkv_w^T + qkv_b  (M=8192, N=2304, K=768), bf16 in/out
// Wave tile 32x64: 2 A-frags amortize 4 B-frags -> 8 WMMA per k-chunk.
// ---------------------------------------------------------------------------
__global__ void __launch_bounds__(128)
qkv_gemm_kernel(const unsigned short* __restrict__ x,
                const unsigned short* __restrict__ w,
                const float* __restrict__ bias, unsigned short* __restrict__ qkv) {
  const int wave = threadIdx.x >> 5;
  const int lane = threadIdx.x & 31;
  const int ln = lane & 15, lh = lane >> 4;
  const int m0 = (blockIdx.x / 9) * 32;
  const int n0 = (blockIdx.x % 9) * 256 + wave * 64;

  v8f acc[2][4] = {};
  const unsigned short* xr0 = x + (size_t)(m0 + ln) * DD;
  const unsigned short* xr1 = x + (size_t)(m0 + 16 + ln) * DD;
  for (int kc = 0; kc < DD; kc += 32) {
    const int o0 = kc + lh * 8;
    v16bf a0 = frag_from_bf16(xr0 + o0, xr0 + o0 + 16);
    v16bf a1 = frag_from_bf16(xr1 + o0, xr1 + o0 + 16);
#pragma unroll
    for (int j = 0; j < 4; ++j) {
      const unsigned short* wr = w + (size_t)(n0 + j * 16 + ln) * DD;  // B[k,n]=W[n,k]
      v16bf bf = frag_from_bf16(wr + o0, wr + o0 + 16);
      acc[0][j] = wmma_bf16(a0, bf, acc[0][j]);
      acc[1][j] = wmma_bf16(a1, bf, acc[1][j]);
    }
  }
#pragma unroll
  for (int t = 0; t < 2; ++t)
#pragma unroll
    for (int j = 0; j < 4; ++j) {
      const int c = n0 + j * 16 + ln;
      const float bb = bias[c];
#pragma unroll
      for (int r = 0; r < 8; ++r) {
        const int m = m0 + t * 16 + lh * 8 + r;                  // f32 C/D layout
        qkv[(size_t)m * TD + c] = f32_to_bf16(acc[t][j][r] + bb);
      }
    }
}

// ---------------------------------------------------------------------------
// Kernel 2: RoPE on q,k + scatter to [B,H,N,HD]; V transposed to [B,H,HD,N]
// ---------------------------------------------------------------------------
__global__ void rope_scatter_kernel(const unsigned short* __restrict__ qkv,
                                    const float* __restrict__ cosp,
                                    const float* __restrict__ sinp,
                                    unsigned short* __restrict__ qh,
                                    unsigned short* __restrict__ kh,
                                    unsigned short* __restrict__ vt) {
  int t = blockIdx.x * blockDim.x + threadIdx.x;   // over B*N*H*HD
  int hd = t & 63;
  int rest = t >> 6;          // (b*N+n)*H + h
  int h = rest % HH;
  int row = rest / HH;        // b*N + n
  int n = row & (NN - 1);
  int b = row >> 10;

  const unsigned short* base = qkv + (size_t)row * TD + h * HD;
  float c = cosp[n * HD + hd];
  float s = sinp[n * HD + hd];
  int prt; float sgn;
  if (hd < 32) { prt = 2 * hd + 1;    sgn = -1.f; }  // rotated = [-t_odd, t_even]
  else         { prt = 2 * (hd - 32); sgn = 1.f; }

  float qv = bf16_to_f32(base[hd]),      qp = bf16_to_f32(base[prt]);
  float kv = bf16_to_f32(base[DD + hd]), kp = bf16_to_f32(base[DD + prt]);

  size_t o = (size_t)((b * HH + h) * NN + n) * HD + hd;
  qh[o] = f32_to_bf16(qv * c + sgn * qp * s);
  kh[o] = f32_to_bf16(kv * c + sgn * kp * s);
  vt[(size_t)((b * HH + h) * HD + hd) * NN + n] = base[2 * DD + hd];
}

// ---------------------------------------------------------------------------
// Kernel 3: flash attention. Block = 4 waves = 64 queries of one (b,h).
// ---------------------------------------------------------------------------
__global__ void __launch_bounds__(128)
attn_kernel(const unsigned short* __restrict__ qh,
            const unsigned short* __restrict__ kh,
            const unsigned short* __restrict__ vt,
            unsigned short* __restrict__ ao) {
  __shared__ __align__(16) unsigned short lK[64 * 64];   // [key][hd]
  __shared__ __align__(16) unsigned short lV[64 * 64];   // [hd][key]
  __shared__ __align__(16) unsigned short lP[4][16 * 32];

  const int tid = threadIdx.x;
  const int wave = tid >> 5;
  const int lane = tid & 31;
  const int ln = lane & 15, lh = lane >> 4;

  const int qblk = blockIdx.x & 15;
  const int h = (blockIdx.x >> 4) % HH;
  const int b = blockIdx.x / (16 * HH);
  const int q0 = qblk * 64 + wave * 16;

  const unsigned short* kbase = kh + (size_t)((b * HH + h) * NN) * HD;
  const unsigned short* vbase = vt + (size_t)((b * HH + h) * HD) * NN;

  // Q fragments (A layout): half0 holds hd {0..7,16..23}, half1 {8..15,24..31}
  const unsigned short* qrow = qh + (size_t)((b * HH + h) * NN + q0 + ln) * HD;
  v16bf aq0 = frag_from_bf16(qrow + lh * 8,      qrow + 16 + lh * 8);
  v16bf aq1 = frag_from_bf16(qrow + 32 + lh * 8, qrow + 48 + lh * 8);

  v8f acc[4] = {};
  float mrow[8], lrow[8];
#pragma unroll
  for (int r = 0; r < 8; ++r) { mrow[r] = -1e30f; lrow[r] = 0.f; }

  const int rr = tid >> 1;          // cooperative-load row 0..63
  const int off = (tid & 1) * 32;

  for (int kb = 0; kb < NN / 64; ++kb) {
    __syncthreads();
    {
      const unsigned short* ks = kbase + (size_t)(kb * 64 + rr) * HD + off;
      const unsigned short* vs = vbase + (size_t)rr * NN + kb * 64 + off;
#pragma unroll
      for (int q = 0; q < 4; ++q) {
        *(uint4*)&lK[rr * 64 + off + q * 8] = *(const uint4*)(ks + q * 8);
        *(uint4*)&lV[rr * 64 + off + q * 8] = *(const uint4*)(vs + q * 8);
      }
      if (kb + 1 < NN / 64) {       // global_prefetch_b8 for next block
        __builtin_prefetch(kbase + (size_t)((kb + 1) * 64 + rr) * HD + off, 0, 0);
        __builtin_prefetch(vbase + (size_t)rr * NN + (kb + 1) * 64 + off, 0, 0);
      }
    }
    __syncthreads();

#pragma unroll
    for (int halfc = 0; halfc < 2; ++halfc) {
      const int c0 = halfc * 32;
      // ---- S = (Q @ K^T) * scale : two 16x16 tiles over 32 keys
      v8f s0 = {}, s1 = {};
      {
        const unsigned short* k0 = &lK[(c0 + ln) * 64];        // key = c0+ln
        const unsigned short* k1 = &lK[(c0 + 16 + ln) * 64];   // key = c0+16+ln
        s0 = wmma_bf16(aq0, frag_from_bf16(k0 + lh * 8, k0 + 16 + lh * 8), s0);
        s0 = wmma_bf16(aq1, frag_from_bf16(k0 + 32 + lh * 8, k0 + 48 + lh * 8), s0);
        s1 = wmma_bf16(aq0, frag_from_bf16(k1 + lh * 8, k1 + 16 + lh * 8), s1);
        s1 = wmma_bf16(aq1, frag_from_bf16(k1 + 32 + lh * 8, k1 + 48 + lh * 8), s1);
      }
      // ---- online softmax (row m = lh*8 + r lives in lanes of this half)
      float p0[8], p1[8];
#pragma unroll
      for (int r = 0; r < 8; ++r) {
        float a0 = s0[r] * 0.125f;            // HD^-0.5
        float a1 = s1[r] * 0.125f;
        float mx = fmaxf(a0, a1);
        mx = fmaxf(mx, __shfl_xor(mx, 1, 32));
        mx = fmaxf(mx, __shfl_xor(mx, 2, 32));
        mx = fmaxf(mx, __shfl_xor(mx, 4, 32));
        mx = fmaxf(mx, __shfl_xor(mx, 8, 32));
        float nm = fmaxf(mrow[r], mx);
        float alpha = __expf(mrow[r] - nm);
        float e0 = __expf(a0 - nm);
        float e1 = __expf(a1 - nm);
        float rs = e0 + e1;
        rs += __shfl_xor(rs, 1, 32);
        rs += __shfl_xor(rs, 2, 32);
        rs += __shfl_xor(rs, 4, 32);
        rs += __shfl_xor(rs, 8, 32);
        lrow[r] = lrow[r] * alpha + rs;
        mrow[r] = nm;
#pragma unroll
        for (int j = 0; j < 4; ++j) acc[j][r] *= alpha;
        p0[r] = e0; p1[r] = e1;
      }
      // ---- P: D-layout -> LDS -> A-layout (per-wave tile, same-wave ordering)
#pragma unroll
      for (int r = 0; r < 8; ++r) {
        int m = lh * 8 + r;
        lP[wave][m * 32 + ln]      = f32_to_bf16(p0[r]);
        lP[wave][m * 32 + 16 + ln] = f32_to_bf16(p1[r]);
      }
      asm volatile("s_wait_dscnt 0" ::: "memory");
      v16bf ap = frag_from_bf16(&lP[wave][ln * 32 + lh * 8],
                                &lP[wave][ln * 32 + 16 + lh * 8]);
      // ---- O += P @ V  (V columns contiguous thanks to transposed lV)
#pragma unroll
      for (int j = 0; j < 4; ++j) {
        const unsigned short* vr = &lV[(j * 16 + ln) * 64 + c0];
        acc[j] = wmma_bf16(ap, frag_from_bf16(vr + lh * 8, vr + 16 + lh * 8),
                           acc[j]);
      }
    }
  }

  // finalize: divide by row sums, store bf16 in [B,N,H,HD] (== [B,N,D])
#pragma unroll
  for (int r = 0; r < 8; ++r) {
    float inv = 1.0f / lrow[r];
    int m = q0 + lh * 8 + r;
#pragma unroll
    for (int j = 0; j < 4; ++j) {
      ao[(size_t)(b * NN + m) * DD + h * HD + j * 16 + ln] =
          f32_to_bf16(acc[j][r] * inv);
    }
  }
}

// ---------------------------------------------------------------------------
// Kernel 4: out = ao @ proj_w^T + proj_b  (M=8192, N=768, K=768), f32 out
// ---------------------------------------------------------------------------
__global__ void __launch_bounds__(128)
proj_gemm_kernel(const unsigned short* __restrict__ a,
                 const unsigned short* __restrict__ w,
                 const float* __restrict__ bias, float* __restrict__ out) {
  const int wave = threadIdx.x >> 5;
  const int lane = threadIdx.x & 31;
  const int ln = lane & 15, lh = lane >> 4;
  const int m0 = (blockIdx.x / 3) * 32;
  const int n0 = (blockIdx.x % 3) * 256 + wave * 64;

  v8f acc[2][4] = {};
  const unsigned short* ar0 = a + (size_t)(m0 + ln) * DD;
  const unsigned short* ar1 = a + (size_t)(m0 + 16 + ln) * DD;
  for (int kc = 0; kc < DD; kc += 32) {
    const int o0 = kc + lh * 8;
    v16bf a0 = frag_from_bf16(ar0 + o0, ar0 + o0 + 16);
    v16bf a1 = frag_from_bf16(ar1 + o0, ar1 + o0 + 16);
#pragma unroll
    for (int j = 0; j < 4; ++j) {
      const unsigned short* wr = w + (size_t)(n0 + j * 16 + ln) * DD;
      v16bf bf = frag_from_bf16(wr + o0, wr + o0 + 16);
      acc[0][j] = wmma_bf16(a0, bf, acc[0][j]);
      acc[1][j] = wmma_bf16(a1, bf, acc[1][j]);
    }
  }
#pragma unroll
  for (int t = 0; t < 2; ++t)
#pragma unroll
    for (int j = 0; j < 4; ++j) {
      const int c = n0 + j * 16 + ln;
      const float bb = bias[c];
#pragma unroll
      for (int r = 0; r < 8; ++r) {
        out[(size_t)(m0 + t * 16 + lh * 8 + r) * DD + c] = acc[t][j][r] + bb;
      }
    }
}

// ---------------------------------------------------------------------------
extern "C" void kernel_launch(void* const* d_in, const int* in_sizes, int n_in,
                              void* d_out, int out_size, void* d_ws, size_t ws_size,
                              hipStream_t stream) {
  const float* x      = (const float*)d_in[0];
  const float* qkv_w  = (const float*)d_in[1];
  const float* qkv_b  = (const float*)d_in[2];
  const float* proj_w = (const float*)d_in[3];
  const float* proj_b = (const float*)d_in[4];
  const float* cosp   = (const float*)d_in[5];
  const float* sinp   = (const float*)d_in[6];
  (void)in_sizes; (void)n_in; (void)out_size; (void)ws_size;

  const size_t n_x  = (size_t)BB * NN * DD;     // 6291456
  const size_t n_w  = (size_t)TD * DD;          // 1769472
  const size_t n_pw = (size_t)DD * DD;          // 589824

  unsigned short* qkv = (unsigned short*)d_ws;                   // 8192*2304 bf16
  unsigned short* qh  = qkv + (size_t)BB * NN * TD;              // [B,H,N,HD]
  unsigned short* kh  = qh + (size_t)BB * HH * NN * HD;
  unsigned short* kv_ = kh;                                      // alias helper
  unsigned short* vt  = kh + (size_t)BB * HH * NN * HD;          // [B,H,HD,N]
  unsigned short* xbf = vt + (size_t)BB * HH * NN * HD;
  unsigned short* wbf = xbf + n_x;
  unsigned short* pwbf = wbf + n_w;
  unsigned short* ao  = qkv;   // reuse: qkv is dead after rope_scatter
  (void)kv_;

  cvt_bf16_kernel<<<dim3((unsigned)(n_x / 4 + 255) / 256), dim3(256), 0, stream>>>(
      x, xbf, (int)(n_x / 4));
  cvt_bf16_kernel<<<dim3((unsigned)(n_w / 4 + 255) / 256), dim3(256), 0, stream>>>(
      qkv_w, wbf, (int)(n_w / 4));
  cvt_bf16_kernel<<<dim3((unsigned)(n_pw / 4 + 255) / 256), dim3(256), 0, stream>>>(
      proj_w, pwbf, (int)(n_pw / 4));

  qkv_gemm_kernel<<<dim3((BB * NN / 32) * 9), dim3(128), 0, stream>>>(
      xbf, wbf, qkv_b, qkv);
  rope_scatter_kernel<<<dim3((BB * NN * HH * HD) / 256), dim3(256), 0, stream>>>(
      qkv, cosp, sinp, qh, kh, vt);
  attn_kernel<<<dim3(BB * HH * (NN / 64)), dim3(128), 0, stream>>>(
      qh, kh, vt, ao);
  proj_gemm_kernel<<<dim3((BB * NN / 32) * 3), dim3(128), 0, stream>>>(
      ao, pwbf, proj_b, (float*)d_out);
}
